// NANOGPT_71098888618735
// MI455X (gfx1250) — matrix operand order repaired and checked by reading, not compiled, not run on hardware
//
#include <hip/hip_runtime.h>
#include <hip/hip_bf16.h>

// ---------------- model constants ----------------
#define BB   4
#define SS   1024
#define DD   768
#define HH   12
#define HSZ  64
#define LL   6
#define VV   50257
#define MM   (BB*SS)          // 4096 activation rows
#define FF4  (4*DD)           // 3072

typedef _Float16 half16 __attribute__((ext_vector_type(16)));
typedef _Float16 half8  __attribute__((ext_vector_type(8)));
typedef _Float16 half4  __attribute__((ext_vector_type(4)));
typedef float    float8 __attribute__((ext_vector_type(8)));
typedef float    floatv4 __attribute__((ext_vector_type(4)));

// ---------------- CDNA5 WMMA ----------------
__device__ __forceinline__ float8 wmma_f16(half16 a, half16 b, float8 c) {
  return __builtin_amdgcn_wmma_f32_16x16x32_f16(false, a, false, b, (short)0, c, false, false);
}

// A fragment: 16x32 (MxK) from row-major LDS tile (ld in halves).
// lane l: m=l&15 ; K groups {0..7,16..23} (lanes 0-15) / {8..15,24..31} (lanes 16-31)
__device__ __forceinline__ half16 load_frag_a(const _Float16* base, int ld, int lane) {
  int m   = lane & 15;
  int off = (lane >> 4) * 8;
  half8 a0 = *(const half8*)(base + m * ld + off);
  half8 a1 = *(const half8*)(base + m * ld + off + 16);
  half16 r;
#pragma unroll
  for (int j = 0; j < 8; ++j) { r[j] = a0[j]; r[j + 8] = a1[j]; }
  return r;
}

// B fragment: 32x16 (KxN) from LDS stored [n][k] (k contiguous, ld halves).
// lane l: n=l&15 ; K groups {0..7,16..23} (lanes 0-15) / {8..15,24..31} (lanes 16-31)
__device__ __forceinline__ half16 load_frag_b(const _Float16* base, int ld, int lane) {
  int n   = lane & 15;
  int off = (lane >> 4) * 8;
  half8 b0 = *(const half8*)(base + n * ld + off);
  half8 b1 = *(const half8*)(base + n * ld + off + 16);
  half16 r;
#pragma unroll
  for (int j = 0; j < 8; ++j) { r[j] = b0[j]; r[j + 8] = b1[j]; }
  return r;
}

// ---------------- CDNA5 async global->LDS (ASYNCcnt DMA) ----------------
__device__ __forceinline__ void async_b128(uint32_t lds_addr, const void* gaddr) {
  asm volatile("global_load_async_to_lds_b128 %0, %1, off"
               :: "v"(lds_addr), "v"(gaddr) : "memory");
}
__device__ __forceinline__ void wait_async0() {
  asm volatile("s_wait_asynccnt 0" ::: "memory");
}
__device__ __forceinline__ uint32_t lds_addr_of(const void* p) {
  return (uint32_t)(uintptr_t)p;   // generic LDS pointer: low 32 bits = LDS offset
}

// ---------------- embedding (fp32 residual stream) ----------------
__global__ void embed_kernel(const int* __restrict__ idx,
                             const float* __restrict__ tok,
                             const float* __restrict__ pos,
                             float* __restrict__ x) {
  long i = (long)blockIdx.x * blockDim.x + threadIdx.x;
  if (i >= (long)MM * DD) return;
  int m = (int)(i / DD), d = (int)(i % DD);
  int t = idx[m];
  x[i] = tok[(long)t * DD + d] + pos[(long)(m & (SS - 1)) * DD + d];
}

// ---------------- layernorm: fp32 in -> f16 out (one wave / row) ----------
__global__ void ln_kernel(const float* __restrict__ x,
                          const float* __restrict__ g,
                          const float* __restrict__ b,
                          _Float16* __restrict__ out) {
  int row  = blockIdx.x * 8 + (threadIdx.x >> 5);
  int lane = threadIdx.x & 31;
  const float* xr = x + (long)row * DD;
  float s = 0.f, s2 = 0.f;
  for (int i = lane; i < DD; i += 32) { float v = xr[i]; s += v; s2 += v * v; }
#pragma unroll
  for (int d = 16; d >= 1; d >>= 1) { s += __shfl_xor(s, d, 32); s2 += __shfl_xor(s2, d, 32); }
  float mu  = s * (1.f / DD);
  float var = s2 * (1.f / DD) - mu * mu;
  float rs  = rsqrtf(var + 1e-5f);
  _Float16* orow = out + (long)row * DD;
  for (int i = lane; i < DD; i += 32)
    orow[i] = (_Float16)((xr[i] - mu) * rs * g[i] + b[i]);
}

// ---------------- WMMA GEMM: 128(M) x 64(N) tile, 4 waves, double-buffered --
// A: f16 row-major [M,K] (lda halves), async-DMA'd to LDS. B: fp32, converted.
//   TRANS=0 : B addr = k*ldb + blockIdx.x*bBlkStride + n_local   (row-major / head-blocked)
//   TRANS=1 : B addr = (n0+n)*ldb + k                            (B = W^T, LM head)
// OUTMODE 0: C[row*ldc+col] ; OUTMODE 1: scatter [B,H,S,HS] (QKV).
template<bool TRANS, int OUTMODE, bool GELU, bool F16OUT>
__global__ __launch_bounds__(128)
void gemm_kernel(const _Float16* __restrict__ A, int lda,
                 const float* __restrict__ B, int ldb, long bBlkStride,
                 const float* __restrict__ bias, const float* __restrict__ resid,
                 void* __restrict__ C, int ldc, int N, int K) {
  __shared__ __align__(64) _Float16 At[2][128 * 32];   // [m][k]
  __shared__ __align__(64) _Float16 Bt[2][64 * 32];    // [n][k]
  const int tid  = threadIdx.x;
  const int wave = tid >> 5, lane = tid & 31;
  const int m0 = blockIdx.y * 128, n0 = blockIdx.x * 64;

  float8 acc[2][4];
#pragma unroll
  for (int mf = 0; mf < 2; ++mf)
#pragma unroll
    for (int f = 0; f < 4; ++f)
#pragma unroll
      for (int r = 0; r < 8; ++r) acc[mf][f][r] = 0.f;

  auto stage = [&](int k0, int buf) {
    // A tile: raw f16 async DMA (128 rows x 64B)
#pragma unroll
    for (int i = 0; i < 4; ++i) {
      int c = tid + i * 128;            // 512 chunks of 16B
      int r = c >> 2, q = c & 3;
      async_b128(lds_addr_of(&At[buf][r * 32 + q * 8]),
                 (const void*)(A + (long)(m0 + r) * lda + k0 + q * 8));
    }
    // B tile: fp32 -> f16 into [n][k]
    if (TRANS) {
#pragma unroll
      for (int i = 0; i < 4; ++i) {
        int c = tid + i * 128;          // 64 rows x 8 float4
        int n = c >> 3, q = c & 7;
        int gn = n0 + n;
        floatv4 fv;
        if (gn < N) fv = *(const floatv4*)(B + (long)gn * ldb + k0 + q * 4);
        else { fv[0] = fv[1] = fv[2] = fv[3] = 0.f; }
        half4 hv;
#pragma unroll
        for (int j = 0; j < 4; ++j) hv[j] = (_Float16)fv[j];
        *(half4*)(&Bt[buf][n * 32 + q * 4]) = hv;
      }
    } else {
#pragma unroll
      for (int i = 0; i < 16; ++i) {
        int e = tid + i * 128;          // 2048 elems
        int n = e >> 5, k = e & 31;
        Bt[buf][n * 32 + k] =
            (_Float16)B[(long)(k0 + k) * ldb + (long)blockIdx.x * bBlkStride + n];
      }
    }
  };

  stage(0, 0);
  wait_async0();
  __syncthreads();

  int cur = 0;
  for (int k0 = 0; k0 < K; k0 += 32) {
    const bool more = (k0 + 32) < K;
    if (more) stage(k0 + 32, cur ^ 1);   // DMA next tile while computing this one

    half16 a0 = load_frag_a(&At[cur][(wave * 32) * 32], 32, lane);
    half16 a1 = load_frag_a(&At[cur][(wave * 32 + 16) * 32], 32, lane);
#pragma unroll
    for (int f = 0; f < 4; ++f) {
      half16 bb = load_frag_b(&Bt[cur][f * 16 * 32], 32, lane);
      acc[0][f] = wmma_f16(a0, bb, acc[0][f]);
      acc[1][f] = wmma_f16(a1, bb, acc[1][f]);
    }
    if (more) { wait_async0(); __syncthreads(); cur ^= 1; }
  }

  const int rbase = (lane >> 4) << 3, c0 = lane & 15;
#pragma unroll
  for (int mf = 0; mf < 2; ++mf)
#pragma unroll
    for (int f = 0; f < 4; ++f)
#pragma unroll
      for (int r = 0; r < 8; ++r) {
        int row = m0 + wave * 32 + mf * 16 + r + rbase;
        int col = n0 + f * 16 + c0;
        if (col < N) {
          float v = acc[mf][f][r];
          if (bias)  v += bias[col];
          if (resid) v += resid[(long)row * ldc + col];
          if (GELU)  v = 0.5f * v * (1.f + erff(v * 0.70710678118f));
          long oaddr;
          if (OUTMODE == 1) {
            int bb = row >> 10, s = row & (SS - 1), hh = col >> 6, hs = col & 63;
            oaddr = (((long)(bb * HH + hh) * SS) + s) * HSZ + hs;
          } else {
            oaddr = (long)row * ldc + col;
          }
          if (F16OUT) ((_Float16*)C)[oaddr] = (_Float16)v;
          else        ((float*)C)[oaddr]    = v;
        }
      }
}

// ---------------- causal flash attention, double-buffered K/V ---------------
// grid (S/64, H, B), block 128 (4 waves x 16 rows). q,k,v: f16 [B,H,S,HS]; out f16 [M, D].
__global__ __launch_bounds__(128)
void attn_kernel(const _Float16* __restrict__ q, const _Float16* __restrict__ k,
                 const _Float16* __restrict__ v, _Float16* __restrict__ out) {
  const int tileM = blockIdx.x, h = blockIdx.y, b = blockIdx.z;
  const int tid = threadIdx.x, wave = tid >> 5, lane = tid & 31;
  const int m0 = tileM * 64;
  const long baseBH = ((long)(b * HH + h)) * SS * HSZ;
  const _Float16* qb = q + baseBH;
  const _Float16* kb = k + baseBH;
  const _Float16* vb = v + baseBH;

  __shared__ __align__(64) _Float16 qt[64 * 64];        // [row][hs]
  __shared__ __align__(64) _Float16 kt[2][64 * 64];     // [key][hs] == B-frag layout for QK^T
  __shared__ __align__(64) _Float16 vt[2][64 * 64];     // [hs][key] == B-frag layout for PV
  __shared__ __align__(64) float    stile[4][16 * 64];
  __shared__ __align__(64) _Float16 ptile[4][16 * 64];
  __shared__ float rowM[4][16], rowL[4][16], rowF[4][16];

  auto stageKV = [&](int t, int buf) {
#pragma unroll
    for (int i = 0; i < 4; ++i) {       // K tile: async raw copy (64 rows x 128B)
      int c = tid + i * 128;
      int r = c >> 3, qk = c & 7;
      async_b128(lds_addr_of(&kt[buf][r * 64 + qk * 8]),
                 (const void*)(kb + (long)(t * 64 + r) * HSZ + qk * 8));
    }
#pragma unroll
    for (int i = 0; i < 4; ++i) {       // V tile: vector load + transposed stores
      int c = tid + i * 128;
      int r = c >> 3, qk = c & 7;
      half8 hv = *(const half8*)(vb + (long)(t * 64 + r) * HSZ + qk * 8);
#pragma unroll
      for (int j = 0; j < 8; ++j) vt[buf][(qk * 8 + j) * 64 + r] = hv[j];
    }
  };

  // async-DMA the q tile (64 rows x 128B) + first K/V tile
#pragma unroll
  for (int i = 0; i < 4; ++i) {
    int c = tid + i * 128;
    int r = c >> 3, qk = c & 7;
    async_b128(lds_addr_of(&qt[r * 64 + qk * 8]),
               (const void*)(qb + (long)(m0 + r) * HSZ + qk * 8));
  }
  stageKV(0, 0);
  if (lane < 16) { rowM[wave][lane] = -__builtin_inff(); rowL[wave][lane] = 0.f; }
  wait_async0();
  __syncthreads();

  half16 aq0 = load_frag_a(qt + wave * 16 * 64, 64, lane);
  half16 aq1 = load_frag_a(qt + wave * 16 * 64 + 32, 64, lane);

  float8 acc[4];
#pragma unroll
  for (int f = 0; f < 4; ++f)
#pragma unroll
    for (int r = 0; r < 8; ++r) acc[f][r] = 0.f;

  const int rbase = (lane >> 4) << 3, c0 = lane & 15;
  const int tmax = (m0 + 63) >> 6;

  int cur = 0;
  for (int t = 0; t <= tmax; ++t) {
    const bool more = t < tmax;
    if (more) stageKV(t + 1, cur ^ 1);  // DMA next K/V while computing this tile

    // scores (16x64 per wave) = Q @ K^T, 1/sqrt(64) applied in f32
    float8 sc[4];
#pragma unroll
    for (int f = 0; f < 4; ++f) {
#pragma unroll
      for (int r = 0; r < 8; ++r) sc[f][r] = 0.f;
      half16 b0 = load_frag_b(&kt[cur][f * 16 * 64], 64, lane);
      sc[f] = wmma_f16(aq0, b0, sc[f]);
      half16 b1 = load_frag_b(&kt[cur][f * 16 * 64 + 32], 64, lane);
      sc[f] = wmma_f16(aq1, b1, sc[f]);
    }
#pragma unroll
    for (int f = 0; f < 4; ++f)
#pragma unroll
      for (int r = 0; r < 8; ++r) {
        int rl = r + rbase;
        int gcol = t * 64 + f * 16 + c0;
        int grow = m0 + wave * 16 + rl;
        stile[wave][rl * 64 + f * 16 + c0] =
            (gcol > grow) ? -__builtin_inff() : sc[f][r] * 0.125f;
      }
    // online softmax, wave-serial over its 16 rows
    for (int i = 0; i < 16; ++i) {
      float x0 = stile[wave][i * 64 + lane];
      float x1 = stile[wave][i * 64 + 32 + lane];
      float mx = fmaxf(x0, x1);
#pragma unroll
      for (int d = 16; d >= 1; d >>= 1) mx = fmaxf(mx, __shfl_xor(mx, d, 32));
      float mold = rowM[wave][i];
      float mnew = fmaxf(mold, mx);
      float p0 = expf(x0 - mnew), p1 = expf(x1 - mnew);
      float ps = p0 + p1;
#pragma unroll
      for (int d = 16; d >= 1; d >>= 1) ps += __shfl_xor(ps, d, 32);
      float fac = (mold == -__builtin_inff()) ? 0.f : expf(mold - mnew);
      if (lane == 0) {
        rowM[wave][i] = mnew;
        rowL[wave][i] = rowL[wave][i] * fac + ps;
        rowF[wave][i] = fac;
      }
      ptile[wave][i * 64 + lane]      = (_Float16)p0;
      ptile[wave][i * 64 + 32 + lane] = (_Float16)p1;
    }
#pragma unroll
    for (int f = 0; f < 4; ++f)
#pragma unroll
      for (int r = 0; r < 8; ++r) acc[f][r] *= rowF[wave][r + rbase];
    // acc += P @ V
#pragma unroll
    for (int s2 = 0; s2 < 2; ++s2) {
      half16 ap = load_frag_a(ptile[wave] + s2 * 32, 64, lane);
#pragma unroll
      for (int f = 0; f < 4; ++f) {
        half16 bv = load_frag_b(&vt[cur][f * 16 * 64 + s2 * 32], 64, lane);
        acc[f] = wmma_f16(ap, bv, acc[f]);
      }
    }
    if (more) { wait_async0(); __syncthreads(); cur ^= 1; }
  }

  // epilogue -> f16 [M, H*HS] row-major
#pragma unroll
  for (int f = 0; f < 4; ++f)
#pragma unroll
    for (int r = 0; r < 8; ++r) {
      int rl = r + rbase;
      float invl = 1.f / rowL[wave][rl];
      int row = m0 + wave * 16 + rl;
      int col = h * HSZ + f * 16 + c0;
      out[(long)row * DD + col] = (_Float16)(acc[f][r] * invl);
    }
}

// ---------------- host orchestration ----------------
extern "C" void kernel_launch(void* const* d_in, const int* in_sizes, int n_in,
                              void* d_out, int out_size, void* d_ws, size_t ws_size,
                              hipStream_t stream) {
  (void)in_sizes; (void)n_in; (void)out_size; (void)ws_size;
  const int*   idx     = (const int*)  d_in[0];
  const float* tok_emb = (const float*)d_in[1];
  const float* pos_emb = (const float*)d_in[2];
  const float* Wq      = (const float*)d_in[3];
  const float* bq      = (const float*)d_in[4];
  const float* Wk      = (const float*)d_in[5];
  const float* bk      = (const float*)d_in[6];
  const float* Wv      = (const float*)d_in[7];
  const float* bv      = (const float*)d_in[8];
  const float* Wproj   = (const float*)d_in[9];
  const float* bproj   = (const float*)d_in[10];
  const float* ln1_g   = (const float*)d_in[11];
  const float* ln1_b   = (const float*)d_in[12];
  const float* ln2_g   = (const float*)d_in[13];
  const float* ln2_b   = (const float*)d_in[14];
  const float* Wf1     = (const float*)d_in[15];
  const float* bf1     = (const float*)d_in[16];
  const float* Wf2     = (const float*)d_in[17];
  const float* bf2     = (const float*)d_in[18];
  const float* lnf_g   = (const float*)d_in[19];
  const float* lnf_b   = (const float*)d_in[20];
  const float* Wlm     = (const float*)d_in[21];
  float* logits = (float*)d_out;

  // workspace layout
  char* w = (char*)d_ws;
  float*     x    = (float*)w;               w += (long)MM * DD * sizeof(float);
  _Float16*  hbuf = (_Float16*)w;            w += (long)MM * DD * sizeof(_Float16);
  _Float16*  qb   = (_Float16*)w;            w += (long)MM * DD * sizeof(_Float16);
  _Float16*  kb   = (_Float16*)w;            w += (long)MM * DD * sizeof(_Float16);
  _Float16*  vb_  = (_Float16*)w;            w += (long)MM * DD * sizeof(_Float16);
  _Float16*  attn = (_Float16*)w;            w += (long)MM * DD * sizeof(_Float16);
  _Float16*  ff   = (_Float16*)w;            w += (long)MM * FF4 * sizeof(_Float16);

  const long wLayer = (long)HH * DD * HSZ;   // 768*768
  const long wProjL = (long)DD * DD;
  const long wF1L   = (long)DD * FF4;
  const long wF2L   = (long)FF4 * DD;

  embed_kernel<<<(MM * DD + 255) / 256, 256, 0, stream>>>(idx, tok_emb, pos_emb, x);

  const dim3 gQKV(DD / 64, MM / 128);
  const dim3 gFF1(FF4 / 64, MM / 128);
  const dim3 gLM((VV + 63) / 64, MM / 128);

  for (int l = 0; l < LL; ++l) {
    ln_kernel<<<MM / 8, 256, 0, stream>>>(x, ln1_g + l * DD, ln1_b + l * DD, hbuf);

    gemm_kernel<false, 1, false, true><<<gQKV, 128, 0, stream>>>(
        hbuf, DD, Wq + l * wLayer, HSZ, (long)DD * HSZ,
        bq + l * HH * HSZ, nullptr, qb, DD, DD, DD);
    gemm_kernel<false, 1, false, true><<<gQKV, 128, 0, stream>>>(
        hbuf, DD, Wk + l * wLayer, HSZ, (long)DD * HSZ,
        bk + l * HH * HSZ, nullptr, kb, DD, DD, DD);
    gemm_kernel<false, 1, false, true><<<gQKV, 128, 0, stream>>>(
        hbuf, DD, Wv + l * wLayer, HSZ, (long)DD * HSZ,
        bv + l * HH * HSZ, nullptr, vb_, DD, DD, DD);

    attn_kernel<<<dim3(SS / 64, HH, BB), 128, 0, stream>>>(qb, kb, vb_, attn);

    gemm_kernel<false, 0, false, false><<<gQKV, 128, 0, stream>>>(
        attn, DD, Wproj + l * wProjL, DD, 64L,
        bproj + l * DD, x, x, DD, DD, DD);

    ln_kernel<<<MM / 8, 256, 0, stream>>>(x, ln2_g + l * DD, ln2_b + l * DD, hbuf);

    gemm_kernel<false, 0, true, true><<<gFF1, 128, 0, stream>>>(
        hbuf, DD, Wf1 + l * wF1L, FF4, 64L,
        bf1 + l * FF4, nullptr, ff, FF4, FF4, DD);
    gemm_kernel<false, 0, false, false><<<gQKV, 128, 0, stream>>>(
        ff, FF4, Wf2 + l * wF2L, DD, 64L,
        bf2 + l * DD, x, x, DD, DD, FF4);
  }

  ln_kernel<<<MM / 8, 256, 0, stream>>>(x, lnf_g, lnf_b, hbuf);

  gemm_kernel<true, 0, false, false><<<gLM, 128, 0, stream>>>(
      hbuf, DD, Wlm, DD, 0L, nullptr, nullptr, logits, VV, VV, DD);
}
